// MDTA_41111426957380
// MI455X (gfx1250) — matrix operand order repaired and checked by reading, not compiled, not run on hardware
//
#include <hip/hip_runtime.h>
#include <hip/hip_bf16.h>
#include <math.h>

// ---------------- problem constants ----------------
#define BATCH 8
#define CH    48
#define HH    256
#define WW    256
#define NPIX  (HH*WW)          // 65536
#define HEADS 8
#define HD    6                // head dim
#define OC3   144              // 3*C

typedef __attribute__((ext_vector_type(16))) _Float16 v16h;
typedef __attribute__((ext_vector_type(8)))  _Float16 h8;
typedef __attribute__((ext_vector_type(8)))  float    v8f;
typedef int v4i __attribute__((__vector_size__(16)));

// ---------------- workspace layout (bytes) ----------------
// qkv_pw  : f16 [B][144][N]          150,994,944
// qkv_dw  : f16 [B][144][N]          150,994,944   (ch 0-47=q, 48-95=k, 96-143=v)
// sumsq   : f32 [B][96]              3,072         (q:0-47, k:48-95)
// attnG   : f32 [B][HEADS][16][16]   65,536        (raw gram, padded)
// Mpad    : f16 [B][64][64]          65,536        (proj_w x blockdiag(softmax attn))
// wpad    : f16 [144][64]            18,432        (qkv_w padded K->64, f16)
#define WS_QKV_PW 0
#define WS_QKV_DW ((size_t)150994944)
#define WS_SUMSQ  ((size_t)301989888)
#define WS_ATTNG  ((size_t)301992960)
#define WS_MPAD   ((size_t)302058496)
#define WS_WPAD   ((size_t)302124032)

#if defined(__gfx1250__) && __has_builtin(__builtin_amdgcn_global_load_async_to_lds_b128)
#define HAVE_ASYNC_LDS 1
#else
#define HAVE_ASYNC_LDS 0
#endif

// ================= weight prep: qkv_w f32[144][48] -> f16[144][64] padded =================
__global__ void k_prep_w(const float* __restrict__ qkv_w, _Float16* __restrict__ wpad) {
    int idx = blockIdx.x * 256 + threadIdx.x;
    if (idx < OC3 * 64) {
        int o = idx >> 6, c = idx & 63;
        wpad[idx] = (c < CH) ? (_Float16)qkv_w[o * CH + c] : (_Float16)0.f;
    }
}

// ================= zero accumulators =================
__global__ void k_zero(float* __restrict__ attnG, float* __restrict__ sumsq) {
    int i = blockIdx.x * 256 + threadIdx.x;
    if (i < BATCH * HEADS * 256) attnG[i] = 0.f;
    if (i < BATCH * 96)          sumsq[i] = 0.f;
}

// ================= kernel A: pointwise qkv GEMM via WMMA =================
// out[b][o][n] = sum_c w[o][c] * x[b][c][n];  o<144, c<48 (padded 64), 128-pixel tiles
__global__ void __launch_bounds__(256) k_pointwise(const float* __restrict__ x,
                                                   const _Float16* __restrict__ wpad,
                                                   _Float16* __restrict__ qkv_pw) {
    const int nb    = blockIdx.x;
    const int b     = nb / (NPIX / 128);
    const int n0    = (nb % (NPIX / 128)) * 128;
    const int tid   = threadIdx.x;
    __shared__ _Float16 xs[128][64];       // [pixel][channel] transposed, 16 KB

    // stage x tile: f32 -> f16, transpose into LDS
    for (int idx = tid; idx < CH * 128; idx += 256) {
        int c = idx >> 7, n = idx & 127;
        xs[n][c] = (_Float16)x[((size_t)b * CH + c) * NPIX + n0 + n];
    }
    for (int idx = tid; idx < 16 * 128; idx += 256) {      // zero-pad channels 48..63
        xs[idx & 127][CH + (idx >> 7)] = (_Float16)0.f;
    }
    __syncthreads();

    const int wave = tid >> 5, lane = tid & 31;
    const int m16 = lane & 15, hh2 = lane >> 4;

    // B fragments (x tile), column n = wave*16 + m16, K = channel
    v16h bfrag[2];
#pragma unroll
    for (int ks = 0; ks < 2; ks++) {
        const _Float16* src = &xs[wave * 16 + m16][ks * 32 + hh2 * 16];
        h8 lo = *(const h8*)src;
        h8 hi = *(const h8*)(src + 8);
#pragma unroll
        for (int i = 0; i < 8; i++) { bfrag[ks][i] = lo[i]; bfrag[ks][i + 8] = hi[i]; }
    }

    _Float16* dst = qkv_pw + (size_t)b * OC3 * NPIX;
    const int nc = n0 + wave * 16 + m16;
#pragma unroll
    for (int mt = 0; mt < 9; mt++) {
        v8f acc = {};
#pragma unroll
        for (int ks = 0; ks < 2; ks++) {
            const _Float16* wr = wpad + (size_t)(mt * 16 + m16) * 64 + ks * 32;
            h8 lo = *(const h8*)(wr + hh2 * 16);
            h8 hi = *(const h8*)(wr + hh2 * 16 + 8);
            v16h a;
#pragma unroll
            for (int i = 0; i < 8; i++) { a[i] = lo[i]; a[i + 8] = hi[i]; }
            acc = __builtin_amdgcn_wmma_f32_16x16x32_f16(false, a, false, bfrag[ks],
                                                         (short)0, acc, false, false);
        }
#pragma unroll
        for (int r = 0; r < 8; r++) {
            int o = mt * 16 + r + 8 * hh2;
            dst[(size_t)o * NPIX + nc] = (_Float16)acc[r];
        }
    }
}

// ================= kernel B: depthwise 3x3 (SAME) + sum-of-squares for q,k =================
// one block = one (b, channel o, 4 image rows); 6 rows staged in LDS via async copy
#define DWROWS 4
__global__ void __launch_bounds__(256) k_dw(const _Float16* __restrict__ pw,
                                            const float* __restrict__ dww,
                                            _Float16* __restrict__ out,
                                            float* __restrict__ sumsq) {
    int id = blockIdx.x;
    const int ytile = id % (HH / DWROWS); id /= (HH / DWROWS);
    const int o = id % OC3; const int b = id / OC3;
    const int y0 = ytile * DWROWS;
    const int tid = threadIdx.x;

    __shared__ _Float16 rows[6][WW];          // rows y0-1 .. y0+4, 3 KB
    const _Float16* base = pw + ((size_t)b * OC3 + o) * NPIX;

    {   // stage: wave w (0..5) copies one full row (32 lanes x 16B = 512B)
        const int wv_ = tid >> 5, lane = tid & 31;
        if (wv_ < 6) {
            int gy = y0 - 1 + wv_;
            _Float16* ldst = &rows[wv_][lane * 8];
            if (gy >= 0 && gy < HH) {
                const _Float16* gsrc = base + (size_t)gy * WW + lane * 8;
#if HAVE_ASYNC_LDS
                __builtin_amdgcn_global_load_async_to_lds_b128(
                    (__attribute__((address_space(1))) v4i*)(uintptr_t)gsrc,
                    (__attribute__((address_space(3))) v4i*)(uint32_t)(uintptr_t)ldst,
                    0, 0);
#else
                *(h8*)ldst = *(const h8*)gsrc;
#endif
            } else {
                h8 z = {};
                *(h8*)ldst = z;                // SAME zero padding rows
            }
        }
    }
#if HAVE_ASYNC_LDS
#if __has_builtin(__builtin_amdgcn_s_wait_asynccnt)
    __builtin_amdgcn_s_wait_asynccnt(0);
#else
    asm volatile("s_wait_asynccnt 0x0" ::: "memory");
#endif
#endif
    __syncthreads();

    float wv9[9];
#pragma unroll
    for (int j = 0; j < 9; j++) wv9[j] = dww[o * 9 + j];

    const int px = tid;
    float col[6][3];
#pragma unroll
    for (int r = 0; r < 6; r++) {
        col[r][0] = (px > 0)      ? (float)rows[r][px - 1] : 0.f;
        col[r][1] =                 (float)rows[r][px];
        col[r][2] = (px < WW - 1) ? (float)rows[r][px + 1] : 0.f;
    }

    float ssq = 0.f;
    _Float16* obase = out + ((size_t)b * OC3 + o) * NPIX + (size_t)y0 * WW + px;
#pragma unroll
    for (int ry = 0; ry < DWROWS; ry++) {
        float acc = 0.f;
#pragma unroll
        for (int ky = 0; ky < 3; ky++)
#pragma unroll
            for (int kx = 0; kx < 3; kx++)
                acc += wv9[ky * 3 + kx] * col[ry + ky][kx];
        obase[(size_t)ry * WW] = (_Float16)acc;
        ssq += acc * acc;
    }

    // block-reduce ssq for q (o<48) and k (48<=o<96)
    __shared__ float red[256];
    red[tid] = ssq;
    __syncthreads();
    for (int s = 128; s > 0; s >>= 1) {
        if (tid < s) red[tid] += red[tid + s];
        __syncthreads();
    }
    if (tid == 0 && o < 96) atomicAdd(&sumsq[b * 96 + o], red[0]);
}

// ================= kernel C: gram G = Q K^T over N via WMMA =================
#define GSPLIT 8   // blocks per (b,head); 8 waves/block -> N/64 = 1024 px per wave
__global__ void __launch_bounds__(256) k_gram(const _Float16* __restrict__ qkv_dw,
                                              float* __restrict__ attnG) {
    int id = blockIdx.x;
    const int s = id % GSPLIT; id /= GSPLIT;
    const int head = id % HEADS; const int b = id / HEADS;
    const int tid = threadIdx.x, wave = tid >> 5, lane = tid & 31;
    const int m16 = lane & 15, hh2 = lane >> 4;

    const int per_wave = NPIX / (GSPLIT * 8);              // 1024
    const int nbase = (s * 8 + wave) * per_wave;

    const _Float16* qb = qkv_dw + ((size_t)b * OC3 + head * HD) * NPIX;
    const _Float16* kb = qkv_dw + ((size_t)b * OC3 + CH + head * HD) * NPIX;
    const int chc = (m16 < HD) ? m16 : (HD - 1);           // clamp; garbage rows/cols ignored

    const _Float16* qrow = qb + (size_t)chc * NPIX;
    const _Float16* krow = kb + (size_t)chc * NPIX;

    v8f acc = {};
    for (int k0 = 0; k0 < per_wave; k0 += 32) {
        const int nk = nbase + k0;
        if ((k0 & 255) == 0 && k0 + 256 < per_wave) {      // prefetch next 512B per stream
            __builtin_prefetch(qrow + nk + 256, 0, 0);
            __builtin_prefetch(krow + nk + 256, 0, 0);
        }
        // A = Q[16 d x 32 n]
        h8 alo = *(const h8*)(qrow + nk + hh2 * 8);
        h8 ahi = *(const h8*)(qrow + nk + 16 + hh2 * 8);
        v16h a;
#pragma unroll
        for (int i = 0; i < 8; i++) { a[i] = alo[i]; a[i + 8] = ahi[i]; }
        // B = K^T[32 n x 16 e]; col e = lane&15, K rows contiguous in n
        h8 blo = *(const h8*)(krow + nk + hh2 * 8);
        h8 bhi = *(const h8*)(krow + nk + 16 + hh2 * 8);
        v16h bf;
#pragma unroll
        for (int i = 0; i < 8; i++) { bf[i] = blo[i]; bf[i + 8] = bhi[i]; }
        acc = __builtin_amdgcn_wmma_f32_16x16x32_f16(false, a, false, bf,
                                                     (short)0, acc, false, false);
    }
    float* g = attnG + (((size_t)b * HEADS + head) << 8);
#pragma unroll
    for (int r = 0; r < 8; r++) atomicAdd(&g[(r + 8 * hh2) * 16 + m16], acc[r]);
}

// ================= kernel D: normalize + softmax + M_b = proj_w x blockdiag(attn) =================
__global__ void __launch_bounds__(256) k_build_M(const float* __restrict__ attnG,
                                                 const float* __restrict__ sumsq,
                                                 const float* __restrict__ temperature,
                                                 const float* __restrict__ proj_w,
                                                 _Float16* __restrict__ Mpad) {
    const int b = blockIdx.x, tid = threadIdx.x;
    __shared__ float sm[HEADS][HD][HD];

    if (tid < HEADS * HD) {
        const int head = tid / HD, d = tid % HD;
        const float* g = attnG + (((size_t)b * HEADS + head) << 8) + d * 16;
        float qn = fmaxf(sqrtf(sumsq[b * 96 + head * HD + d]), 1e-12f);
        float t = temperature[head];
        float row[HD]; float mx = -1e30f;
#pragma unroll
        for (int e = 0; e < HD; e++) {
            float kn = fmaxf(sqrtf(sumsq[b * 96 + 48 + head * HD + e]), 1e-12f);
            row[e] = g[e] / (qn * kn) * t;
            mx = fmaxf(mx, row[e]);
        }
        float ssum = 0.f;
#pragma unroll
        for (int e = 0; e < HD; e++) { row[e] = expf(row[e] - mx); ssum += row[e]; }
        float inv = 1.f / ssum;
#pragma unroll
        for (int e = 0; e < HD; e++) sm[head][d][e] = row[e] * inv;
    }
    __syncthreads();

    _Float16* Mb = Mpad + (size_t)b * 64 * 64;
    for (int idx = tid; idx < 64 * 64; idx += 256) {
        int o = idx >> 6, cc = idx & 63;
        float v = 0.f;
        if (o < CH && cc < CH) {
            int head = cc / HD, e = cc % HD;
#pragma unroll
            for (int d = 0; d < HD; d++) v += proj_w[o * CH + head * HD + d] * sm[head][d][e];
        }
        Mb[idx] = (_Float16)v;
    }
}

// ================= kernel E: out = M_b x V  (48x48 x [48,N]) via WMMA, f32 output =================
__global__ void __launch_bounds__(256) k_av_proj(const _Float16* __restrict__ qkv_dw,
                                                 const _Float16* __restrict__ Mpad,
                                                 float* __restrict__ out) {
    const int nb  = blockIdx.x;
    const int b   = nb / (NPIX / 128);
    const int n0  = (nb % (NPIX / 128)) * 128;
    const int tid = threadIdx.x;
    __shared__ _Float16 vs[128][64];

    const _Float16* vbase = qkv_dw + ((size_t)b * OC3 + 2 * CH) * NPIX;
    for (int idx = tid; idx < CH * 128; idx += 256) {
        int c = idx >> 7, n = idx & 127;
        vs[n][c] = vbase[(size_t)c * NPIX + n0 + n];
    }
    for (int idx = tid; idx < 16 * 128; idx += 256) {
        vs[idx & 127][CH + (idx >> 7)] = (_Float16)0.f;
    }
    __syncthreads();

    const int wave = tid >> 5, lane = tid & 31;
    const int m16 = lane & 15, hh2 = lane >> 4;

    v16h bfrag[2];
#pragma unroll
    for (int ks = 0; ks < 2; ks++) {
        const _Float16* src = &vs[wave * 16 + m16][ks * 32 + hh2 * 16];
        h8 lo = *(const h8*)src;
        h8 hi = *(const h8*)(src + 8);
#pragma unroll
        for (int i = 0; i < 8; i++) { bfrag[ks][i] = lo[i]; bfrag[ks][i + 8] = hi[i]; }
    }

    const _Float16* Mb = Mpad + (size_t)b * 64 * 64;
    float* dst = out + (size_t)b * CH * NPIX;
    const int nc = n0 + wave * 16 + m16;
#pragma unroll
    for (int mt = 0; mt < 3; mt++) {
        v8f acc = {};
#pragma unroll
        for (int ks = 0; ks < 2; ks++) {
            const _Float16* wr = Mb + (size_t)(mt * 16 + m16) * 64 + ks * 32;
            h8 lo = *(const h8*)(wr + hh2 * 16);
            h8 hi = *(const h8*)(wr + hh2 * 16 + 8);
            v16h a;
#pragma unroll
            for (int i = 0; i < 8; i++) { a[i] = lo[i]; a[i + 8] = hi[i]; }
            acc = __builtin_amdgcn_wmma_f32_16x16x32_f16(false, a, false, bfrag[ks],
                                                         (short)0, acc, false, false);
        }
#pragma unroll
        for (int r = 0; r < 8; r++) {
            int o = mt * 16 + r + 8 * hh2;           // 0..47, all valid
            dst[(size_t)o * NPIX + nc] = acc[r];
        }
    }
}

// ================= host launcher =================
extern "C" void kernel_launch(void* const* d_in, const int* in_sizes, int n_in,
                              void* d_out, int out_size, void* d_ws, size_t ws_size,
                              hipStream_t stream) {
    const float* x      = (const float*)d_in[0];   // [B,48,H,W]
    const float* qkv_w  = (const float*)d_in[1];   // [144,48]
    const float* dw_w   = (const float*)d_in[2];   // [144,1,3,3]
    const float* proj_w = (const float*)d_in[3];   // [48,48]
    const float* temp   = (const float*)d_in[4];   // [8]
    float* outp         = (float*)d_out;

    char* ws = (char*)d_ws;
    _Float16* qkv_pw = (_Float16*)(ws + WS_QKV_PW);
    _Float16* qkv_dw = (_Float16*)(ws + WS_QKV_DW);
    float*    sumsq  = (float*)   (ws + WS_SUMSQ);
    float*    attnG  = (float*)   (ws + WS_ATTNG);
    _Float16* Mpad   = (_Float16*)(ws + WS_MPAD);
    _Float16* wpad   = (_Float16*)(ws + WS_WPAD);

    (void)in_sizes; (void)n_in; (void)out_size; (void)ws_size;

    k_prep_w<<<(OC3 * 64 + 255) / 256, 256, 0, stream>>>(qkv_w, wpad);
    k_zero<<<(BATCH * HEADS * 256 + 255) / 256, 256, 0, stream>>>(attnG, sumsq);
    k_pointwise<<<BATCH * (NPIX / 128), 256, 0, stream>>>(x, wpad, qkv_pw);
    k_dw<<<BATCH * OC3 * (HH / DWROWS), 256, 0, stream>>>(qkv_pw, dw_w, qkv_dw, sumsq);
    k_gram<<<BATCH * HEADS * GSPLIT, 256, 0, stream>>>(qkv_dw, attnG);
    k_build_M<<<BATCH, 256, 0, stream>>>(attnG, sumsq, temp, proj_w, Mpad);
    k_av_proj<<<BATCH * (NPIX / 128), 256, 0, stream>>>(qkv_dw, Mpad, outp);
}
